// LSTM_42399917146236
// MI455X (gfx1250) — compile-verified, matrix operand order checked by reading, not run on hardware
//
#include <hip/hip_runtime.h>

// ---------------------------------------------------------------------------
// Bidirectional 3-layer LSTM (T=65536, H=80) for MI455X (gfx1250, wave32).
//  - Gate-input projections done as bf16 WMMA GEMMs (v_wmma_f32_16x16x32_bf16)
//  - Serial recurrence: 1 block (320 thr) per direction, W_hh rows in VGPRs,
//    h broadcast from LDS, gates prefetched with global_prefetch_b8.
// ---------------------------------------------------------------------------

#define T_SEQ 65536
#define HID   80          // hidden size
#define G4    320         // 4*H gates
#define IN2   160         // 2*H input width of higher layers

typedef __bf16 bf16_t;
typedef bf16_t v16bf __attribute__((ext_vector_type(16)));
typedef float  v8f   __attribute__((ext_vector_type(8)));

union Frag { unsigned int u[8]; v16bf v; };

static __device__ __forceinline__ unsigned short f32_to_bf16(float f) {
    unsigned int u = __float_as_uint(f);
    u += 0x7FFFu + ((u >> 16) & 1u);          // round-to-nearest-even
    return (unsigned short)(u >> 16);
}
static __device__ __forceinline__ float bf16_to_f32(unsigned short h) {
    return __uint_as_float(((unsigned int)h) << 16);
}

// --------------------------- prep kernels ----------------------------------

// Fused bias: biasAll[layer][d][r] = b_ih + b_hh   (layer 0..2)
__global__ void prep_bias(const float* __restrict__ b_ih_l0,
                          const float* __restrict__ b_hh_l0,
                          const float* __restrict__ b_ih_hi,
                          const float* __restrict__ b_hh_hi,
                          float* __restrict__ biasAll) {
    int idx = blockIdx.x * blockDim.x + threadIdx.x;
    if (idx >= 3 * 2 * G4) return;
    int layer = idx / (2 * G4);
    int j     = idx % (2 * G4);               // d*320 + r
    float v;
    if (layer == 0) v = b_ih_l0[j] + b_hh_l0[j];
    else            v = b_ih_hi[(layer - 1) * 2 * G4 + j] +
                        b_hh_hi[(layer - 1) * 2 * G4 + j];
    biasAll[idx] = v;
}

// Pack W_ih_hi (l,d,r=320,k=160) f32 -> bf16 with K-pairs interleaved so a
// single dword load yields the (k,k+1) pair for one B-fragment slot:
//   us[ ((l*2+d)*51200) + ((k>>1)*320 + r)*2 + (k&1) ] = bf16(w)
__global__ void pack_wih(const float* __restrict__ w_ih_hi,
                         unsigned short* __restrict__ wp_us) {
    int idx = blockIdx.x * blockDim.x + threadIdx.x;
    if (idx >= 2 * 2 * G4 * IN2) return;
    int ld = idx / (G4 * IN2);                // (l*2+d)
    int rem = idx % (G4 * IN2);
    int r = rem / IN2;
    int k = rem % IN2;
    float w = w_ih_hi[((size_t)ld * G4 + r) * IN2 + k];
    size_t dst = (size_t)ld * (G4 * IN2) + ((size_t)(k >> 1) * G4 + r) * 2 + (k & 1);
    wp_us[dst] = f32_to_bf16(w);
}

// Layer-0 gate precompute (input width 1): gates[d][s][r] = x[t]*w + bias
__global__ void l0_gates(const float* __restrict__ x,
                         const float* __restrict__ w_ih_l0,   // (2,320,1)
                         const float* __restrict__ biasAll,   // layer 0 slice
                         float* __restrict__ gates) {
    size_t idx = (size_t)blockIdx.x * blockDim.x + threadIdx.x;
    if (idx >= (size_t)2 * T_SEQ * G4) return;
    int d = (int)(idx / ((size_t)T_SEQ * G4));
    size_t rem = idx % ((size_t)T_SEQ * G4);
    int s = (int)(rem / G4);
    int r = (int)(rem % G4);
    int t = d ? (T_SEQ - 1 - s) : s;          // backward scan reads reversed x
    gates[idx] = x[t] * w_ih_l0[d * G4 + r] + biasAll[d * G4 + r];
}

// ------------------- WMMA gate GEMM for higher layers ----------------------
// gates[d][s][0..319] = Xbf[row(s,d)] (1x160 bf16) @ W_ih^T (160x320) + bias
// One wave computes one 16x16 tile; K looped 160/32 = 5 WMMAs.
__global__ __launch_bounds__(256)
void gates_gemm(const unsigned short* __restrict__ xbf,   // T x 160 bf16
                const unsigned int*  __restrict__ wpack,  // per-layer base
                const float*         __restrict__ bias,   // per-layer [2][320]
                float*               __restrict__ gates)  // [2][T][320]
{
    const int wid  = blockIdx.x * 8 + (threadIdx.x >> 5);
    const int lane = threadIdx.x & 31;
    const int half = lane >> 4;
    const int mn   = lane & 15;

    const int tilesPerDir = (T_SEQ / 16) * (G4 / 16);     // 4096*20 = 81920
    if (wid >= 2 * tilesPerDir) return;
    const int d   = wid / tilesPerDir;
    const int rem = wid % tilesPerDir;
    const int tt  = rem / (G4 / 16);
    const int tn  = rem % (G4 / 16);

    const int  row  = tt * 16 + mn;                       // A-matrix M per lane
    const long Xrow = d ? (long)(T_SEQ - 1 - row) : (long)row;
    const unsigned short* xr = xbf + (size_t)Xrow * IN2;
    const unsigned int*   Wp = wpack + (size_t)d * ((IN2 / 2) * G4); // 25600 u32

    v8f acc = {};
    #pragma unroll
    for (int kc = 0; kc < 5; ++kc) {
        Frag a, b;
        #pragma unroll
        for (int v = 0; v < 8; ++v) {
            // ISA 7.12.2 16-bit A layout: K pair per VGPR, +8 for lanes 16-31
            const int k = kc * 32 + ((v < 4) ? 2 * v : 8 + 2 * v) + 8 * half;
            a.u[v] = *reinterpret_cast<const unsigned int*>(xr + k);
            b.u[v] = Wp[(size_t)(k >> 1) * G4 + tn * 16 + mn];
        }
        acc = __builtin_amdgcn_wmma_f32_16x16x32_bf16(
                  false, a.v, false, b.v, (short)0, acc, false, false);
    }

    const int gr = tn * 16 + mn;
    const float bb = bias[d * G4 + gr];
    #pragma unroll
    for (int j = 0; j < 8; ++j) {
        const int gs = tt * 16 + j + 8 * half;            // C/D layout: M=j+8*half
        gates[((size_t)d * T_SEQ + gs) * G4 + gr] = acc[j] + bb;
    }
}

// ------------------------- serial LSTM scan --------------------------------
// grid.x = 2 (direction). 320 threads: thread r owns gate r; threads 0..79
// also own (c,h) of hidden unit r. W_hh row kept in 80 VGPRs.
__global__ __launch_bounds__(320)
void lstm_scan(const float* __restrict__ gates,   // [2][T][320] precomputed
               const float* __restrict__ whh,     // [2][320][80], layer slice
               unsigned short* __restrict__ xout) // T x 160 bf16
{
    const int d = blockIdx.x;
    const int r = threadIdx.x;
    const float* g = gates + (size_t)d * T_SEQ * G4;
    const float* W = whh + ((size_t)d * G4 + r) * HID;

    float w[HID];
    #pragma unroll
    for (int q = 0; q < HID / 4; ++q) {
        float4 t = *reinterpret_cast<const float4*>(W + 4 * q);
        w[4 * q + 0] = t.x; w[4 * q + 1] = t.y;
        w[4 * q + 2] = t.z; w[4 * q + 3] = t.w;
    }

    __shared__ float hsh[HID];
    __shared__ float ga[G4];
    if (r < HID) hsh[r] = 0.0f;
    float cst = 0.0f;
    __syncthreads();

    for (int s = 0; s < T_SEQ; ++s) {
        float acc = g[(size_t)s * G4 + r];
        if (s + 8 < T_SEQ)
            __builtin_prefetch(&g[(size_t)(s + 8) * G4 + r], 0, 0);
        #pragma unroll
        for (int q = 0; q < HID / 4; ++q) {
            float4 hv = *reinterpret_cast<const float4*>(&hsh[4 * q]);
            acc += hv.x * w[4 * q + 0] + hv.y * w[4 * q + 1]
                 + hv.z * w[4 * q + 2] + hv.w * w[4 * q + 3];
        }
        // gate order: i[0..79] f[80..159] g[160..239] o[240..319]
        float av = (r >= 160 && r < 240) ? tanhf(acc)
                                         : 1.0f / (1.0f + expf(-acc));
        ga[r] = av;
        __syncthreads();
        if (r < HID) {
            float iv = ga[r], fv = ga[HID + r], gv = ga[2 * HID + r],
                  ov = ga[3 * HID + r];
            cst = fv * cst + iv * gv;
            float hv = ov * tanhf(cst);
            hsh[r] = hv;
            int row = d ? (T_SEQ - 1 - s) : s;   // un-reverse backward outputs
            xout[(size_t)row * IN2 + d * HID + r] = f32_to_bf16(hv);
        }
        __syncthreads();
    }
}

// ----------------------------- final linear --------------------------------
__global__ void final_linear(const unsigned short* __restrict__ x2,
                             const float* __restrict__ lw,
                             const float* __restrict__ lb,
                             float* __restrict__ out) {
    __shared__ float red[IN2];
    int t = threadIdx.x;
    red[t] = bf16_to_f32(x2[(size_t)(T_SEQ - 1) * IN2 + t]) * lw[t];
    __syncthreads();
    if (t == 0) {
        float s = 0.0f;
        for (int i = 0; i < IN2; ++i) s += red[i];
        out[0] = s + lb[0];
    }
}

// ----------------------------- host driver ---------------------------------
extern "C" void kernel_launch(void* const* d_in, const int* in_sizes, int n_in,
                              void* d_out, int out_size, void* d_ws, size_t ws_size,
                              hipStream_t stream) {
    const float* input_seq = (const float*)d_in[0];
    const float* w_ih_l0   = (const float*)d_in[1];
    const float* w_hh_l0   = (const float*)d_in[2];
    const float* b_ih_l0   = (const float*)d_in[3];
    const float* b_hh_l0   = (const float*)d_in[4];
    const float* w_ih_hi   = (const float*)d_in[5];
    const float* w_hh_hi   = (const float*)d_in[6];
    const float* b_ih_hi   = (const float*)d_in[7];
    const float* b_hh_hi   = (const float*)d_in[8];
    const float* linear_w  = (const float*)d_in[9];
    const float* linear_b  = (const float*)d_in[10];
    float* out = (float*)d_out;

    // Workspace carve-up (~210 MB)
    char* p = (char*)d_ws;
    float* gates = (float*)p;            p += (size_t)2 * T_SEQ * G4 * 4;   // 168 MB
    unsigned short* xA = (unsigned short*)p; p += (size_t)T_SEQ * IN2 * 2;  // 21 MB
    unsigned short* xB = (unsigned short*)p; p += (size_t)T_SEQ * IN2 * 2;  // 21 MB
    unsigned int* wpack = (unsigned int*)p;  p += (size_t)2 * 2 * (IN2/2) * G4 * 4;
    float* biasAll = (float*)p;          p += (size_t)3 * 2 * G4 * 4;
    (void)ws_size;

    const int perLayerU32 = 2 * (IN2 / 2) * G4;  // 51200 u32 per layer (2 dirs)

    // 1) prep
    prep_bias<<<(3 * 2 * G4 + 255) / 256, 256, 0, stream>>>(
        b_ih_l0, b_hh_l0, b_ih_hi, b_hh_hi, biasAll);
    pack_wih<<<(2 * 2 * G4 * IN2 + 255) / 256, 256, 0, stream>>>(
        w_ih_hi, (unsigned short*)wpack);

    // 2) layer 0
    l0_gates<<<(int)(((size_t)2 * T_SEQ * G4 + 255) / 256), 256, 0, stream>>>(
        input_seq, w_ih_l0, biasAll + 0 * 2 * G4, gates);
    lstm_scan<<<2, 320, 0, stream>>>(gates, w_hh_l0, xA);

    // 3) layer 1 (hi layer 0)
    const int gemmBlocks = (2 * (T_SEQ / 16) * (G4 / 16)) / 8;   // 20480
    gates_gemm<<<gemmBlocks, 256, 0, stream>>>(
        xA, wpack + 0 * perLayerU32, biasAll + 1 * 2 * G4, gates);
    lstm_scan<<<2, 320, 0, stream>>>(gates, w_hh_hi + (size_t)0 * 2 * G4 * HID, xB);

    // 4) layer 2 (hi layer 1)
    gates_gemm<<<gemmBlocks, 256, 0, stream>>>(
        xB, wpack + 1 * perLayerU32, biasAll + 2 * 2 * G4, gates);
    lstm_scan<<<2, 320, 0, stream>>>(gates, w_hh_hi + (size_t)1 * 2 * G4 * HID, xA);

    // 5) final projection of x[-1]
    final_linear<<<1, IN2, 0, stream>>>(xA, linear_w, linear_b, out);
}
